// MPNEncoder_44547400794478
// MI455X (gfx1250) — matrix operand order repaired and checked by reading, not compiled, not run on hardware
//
#include <hip/hip_runtime.h>
#include <hip/hip_bf16.h>

// D-MPNN forward for MI455X (gfx1250), bf16 WMMA (v_wmma_f32_16x16x32_bf16)
// with fp32 accumulation. Bandwidth-bound by design (~116 GFLOP vs ~3 GB moved).
// Workspace requirement: ~737 MB.

typedef __attribute__((ext_vector_type(16))) __bf16 v16bf;
typedef __attribute__((ext_vector_type(8)))  __bf16 v8bf;
typedef __attribute__((ext_vector_type(4)))  __bf16 v4bf;
typedef __attribute__((ext_vector_type(8)))  float  v8f;

#define NAq 100000
#define NBq 200000
#define MAXNB 6
#define ATOM_FDIM 133
#define BOND_FDIM 147
#define HDIM 300
#define HPAD 304          // fp32 row stride = 19 tiles * 16
#define NT_VALID 19
#define NT_ALLOC 20       // one extra zero tile so waves process 4 n-tiles blindly
#define KP_BOND 160       // 147 padded to mult of 32
#define KP_H    320       // 300 padded
#define KP_O    448       // 433 padded

// ---------------- helper kernels ----------------

// fp32 [rows,kin] -> bf16 [rows,kpad], zero-padded
__global__ void cvt_pad_bf16(const float* __restrict__ X, __bf16* __restrict__ Y,
                             long rows, int kin, int kpad) {
  long idx = (long)blockIdx.x * blockDim.x + threadIdx.x;
  long total = rows * (long)kpad;
  if (idx >= total) return;
  int k = (int)(idx % kpad);
  long r = idx / kpad;
  float v = (k < kin) ? X[r * kin + k] : 0.0f;
  Y[idx] = (__bf16)v;
}

// W [K,N] fp32 row-major -> per-lane swizzled bf16 B operand:
// layout [ntile][kstep][lane][16 contiguous bf16]; lane<16: N=lane, K=kb..kb+15;
// lane>=16: N=lane-16, K=kb+16..kb+31 (contiguous k order, same as load_a16x32 uses for A).
__global__ void swizzle_w(const float* __restrict__ W, __bf16* __restrict__ Wsw,
                          int K, int N, int ksteps) {
  int idx = blockIdx.x * blockDim.x + threadIdx.x;
  int total = NT_ALLOC * ksteps * 32;
  if (idx >= total) return;
  int lane = idx & 31;
  int ks   = (idx >> 5) % ksteps;
  int nt   = idx / (32 * ksteps);
  int n    = nt * 16 + (lane & 15);
  int kb   = ks * 32 + (lane >> 4) * 16;
  __bf16* out = Wsw + (size_t)idx * 16;
#pragma unroll
  for (int i = 0; i < 16; ++i) {
    int k = kb + i;
    float v = (k < K && n < N) ? W[(size_t)k * N + n] : 0.0f;
    out[i] = (__bf16)v;
  }
}

// a_message[a,:] = sum_n msg[a2b[a,n],:]   (float4-vectorized over HPAD)
__global__ void atom_gather_sum(const float* __restrict__ msg, const int* __restrict__ a2b,
                                float* __restrict__ amsg) {
  int t = blockIdx.x * blockDim.x + threadIdx.x;
  const int CH = HPAD / 4;  // 76
  if (t >= NAq * CH) return;
  int a = t / CH;
  int c = (t - a * CH) * 4;
  float4 s = make_float4(0.f, 0.f, 0.f, 0.f);
#pragma unroll
  for (int n = 0; n < MAXNB; ++n) {
    int b = a2b[a * MAXNB + n];
    const float4 v = *(const float4*)(msg + (size_t)b * HPAD + c);
    s.x += v.x; s.y += v.y; s.z += v.z; s.w += v.w;
  }
  *(float4*)(amsg + (size_t)a * HPAD + c) = s;
}

// pre[b,:] = bf16( amsg[b2a[b],:] - msg[b2revb[b],:] ), zero-padded to KP_H
__global__ void bond_prep(const float* __restrict__ amsg, const float* __restrict__ msg,
                          const int* __restrict__ b2a, const int* __restrict__ b2revb,
                          __bf16* __restrict__ pre) {
  int t = blockIdx.x * blockDim.x + threadIdx.x;
  const int CH = KP_H / 4;  // 80
  if (t >= NBq * CH) return;
  int b = t / CH;
  int k = (t - b * CH) * 4;
  v4bf o;
  if (k < HDIM) {
    int ba = b2a[b];
    int br = b2revb[b];
    const float4 va = *(const float4*)(amsg + (size_t)ba * HPAD + k);
    const float4 vm = *(const float4*)(msg  + (size_t)br * HPAD + k);
    o[0] = (__bf16)(va.x - vm.x);
    o[1] = (__bf16)(va.y - vm.y);
    o[2] = (__bf16)(va.z - vm.z);
    o[3] = (__bf16)(va.w - vm.w);
  } else {
    o[0] = (__bf16)0.f; o[1] = (__bf16)0.f; o[2] = (__bf16)0.f; o[3] = (__bf16)0.f;
  }
  *(v4bf*)(pre + (size_t)b * KP_H + k) = o;
}

// a_input[a,:] = bf16 concat(f_atoms[a,:], amsg[a,0:H]), zero-padded to KP_O
__global__ void build_ainput(const float* __restrict__ f_atoms, const float* __restrict__ amsg,
                             __bf16* __restrict__ ainp) {
  long t = (long)blockIdx.x * blockDim.x + threadIdx.x;
  if (t >= (long)NAq * KP_O) return;
  int k = (int)(t % KP_O);
  long a = t / KP_O;
  float v;
  if (k < ATOM_FDIM)             v = f_atoms[a * ATOM_FDIM + k];
  else if (k < ATOM_FDIM + HDIM) v = amsg[a * HPAD + (k - ATOM_FDIM)];
  else                           v = 0.f;
  ainp[t] = (__bf16)v;
}

// ---------------- WMMA GEMM core ----------------

__device__ __forceinline__ v16bf join8(v8bf lo, v8bf hi) {
  v16bf r;
#pragma unroll
  for (int i = 0; i < 8; ++i) { r[i] = lo[i]; r[i + 8] = hi[i]; }
  return r;
}

// A 16x32 bf16 tile, row-major source with row stride lda.
// lane<16: row=lane, K = 32ks+{0..15}; lane>=16: row=lane-16, K = 32ks+{16..31}.
__device__ __forceinline__ v16bf load_a16x32(const __bf16* __restrict__ A, int lda,
                                             int mtile, int ks, int lane) {
  const __bf16* p = A + (size_t)(mtile * 16 + (lane & 15)) * lda + ks * 32 + ((lane >> 4) << 4);
  v8bf lo = *(const v8bf*)p;
  v8bf hi = *(const v8bf*)(p + 8);
  return join8(lo, hi);
}

// B 32x16 bf16 tile from the pre-swizzled weight buffer (fully coalesced).
__device__ __forceinline__ v16bf load_b32x16(const __bf16* __restrict__ Wsw,
                                             int ntile, int ks, int ksteps, int lane) {
  const __bf16* p = Wsw + ((((size_t)ntile * ksteps + ks) << 5) + lane) * 16;
  v8bf lo = *(const v8bf*)p;
  v8bf hi = *(const v8bf*)(p + 8);
  return join8(lo, hi);
}

// GEMM1: inp = A@W_i ; msg = relu(inp)     (both strided HPAD fp32)
__global__ __launch_bounds__(256)
void gemm_wi(const __bf16* __restrict__ A, const __bf16* __restrict__ Wsw,
             float* __restrict__ inp, float* __restrict__ msg, int mtiles, int ksteps, int lda) {
  int lane = threadIdx.x & 31;
  int mtile = blockIdx.x * 8 + (threadIdx.x >> 5);
  if (mtile >= mtiles) return;
  int nt0 = blockIdx.y * 4;
  v8f c[4];
#pragma unroll
  for (int j = 0; j < 4; ++j)
#pragma unroll
    for (int r = 0; r < 8; ++r) c[j][r] = 0.f;
  for (int ks = 0; ks < ksteps; ++ks) {
    v16bf a = load_a16x32(A, lda, mtile, ks, lane);
#pragma unroll
    for (int j = 0; j < 4; ++j) {
      v16bf b = load_b32x16(Wsw, nt0 + j, ks, ksteps, lane);
      c[j] = __builtin_amdgcn_wmma_f32_16x16x32_bf16(false, a, false, b, (short)0, c[j], false, false);
    }
  }
  int rbase = mtile * 16 + ((lane >> 4) << 3);
  int cbase = lane & 15;
#pragma unroll
  for (int j = 0; j < 4; ++j) {
    int nt = nt0 + j;
    if (nt < NT_VALID) {
      int col = nt * 16 + cbase;
#pragma unroll
      for (int r = 0; r < 8; ++r) {
        float v = c[j][r];
        size_t off = (size_t)(rbase + r) * HPAD + col;
        inp[off] = v;
        msg[off] = v > 0.f ? v : 0.f;
      }
    }
  }
}

// GEMM2: msg = relu(inp + pre@W_h)
__global__ __launch_bounds__(256)
void gemm_wh(const __bf16* __restrict__ A, const __bf16* __restrict__ Wsw,
             const float* __restrict__ inp, float* __restrict__ msg,
             int mtiles, int ksteps, int lda) {
  int lane = threadIdx.x & 31;
  int mtile = blockIdx.x * 8 + (threadIdx.x >> 5);
  if (mtile >= mtiles) return;
  int nt0 = blockIdx.y * 4;
  v8f c[4];
#pragma unroll
  for (int j = 0; j < 4; ++j)
#pragma unroll
    for (int r = 0; r < 8; ++r) c[j][r] = 0.f;
  for (int ks = 0; ks < ksteps; ++ks) {
    v16bf a = load_a16x32(A, lda, mtile, ks, lane);
#pragma unroll
    for (int j = 0; j < 4; ++j) {
      v16bf b = load_b32x16(Wsw, nt0 + j, ks, ksteps, lane);
      c[j] = __builtin_amdgcn_wmma_f32_16x16x32_bf16(false, a, false, b, (short)0, c[j], false, false);
    }
  }
  int rbase = mtile * 16 + ((lane >> 4) << 3);
  int cbase = lane & 15;
#pragma unroll
  for (int j = 0; j < 4; ++j) {
    int nt = nt0 + j;
    if (nt < NT_VALID) {
      int col = nt * 16 + cbase;
#pragma unroll
      for (int r = 0; r < 8; ++r) {
        size_t off = (size_t)(rbase + r) * HPAD + col;
        float v = inp[off] + c[j][r];
        msg[off] = v > 0.f ? v : 0.f;
      }
    }
  }
}

// GEMM3: out = relu(a_input@W_o + b_o), dense [NA, 300] output
__global__ __launch_bounds__(256)
void gemm_wo(const __bf16* __restrict__ A, const __bf16* __restrict__ Wsw,
             const float* __restrict__ bias, float* __restrict__ out,
             int mtiles, int ksteps, int lda) {
  int lane = threadIdx.x & 31;
  int mtile = blockIdx.x * 8 + (threadIdx.x >> 5);
  if (mtile >= mtiles) return;
  int nt0 = blockIdx.y * 4;
  v8f c[4];
#pragma unroll
  for (int j = 0; j < 4; ++j)
#pragma unroll
    for (int r = 0; r < 8; ++r) c[j][r] = 0.f;
  for (int ks = 0; ks < ksteps; ++ks) {
    v16bf a = load_a16x32(A, lda, mtile, ks, lane);
#pragma unroll
    for (int j = 0; j < 4; ++j) {
      v16bf b = load_b32x16(Wsw, nt0 + j, ks, ksteps, lane);
      c[j] = __builtin_amdgcn_wmma_f32_16x16x32_bf16(false, a, false, b, (short)0, c[j], false, false);
    }
  }
  int rbase = mtile * 16 + ((lane >> 4) << 3);
  int cbase = lane & 15;
#pragma unroll
  for (int j = 0; j < 4; ++j) {
    int col = (nt0 + j) * 16 + cbase;
    if (col < HDIM) {
      float bo = bias[col];
#pragma unroll
      for (int r = 0; r < 8; ++r) {
        float v = c[j][r] + bo;
        out[(size_t)(rbase + r) * HDIM + col] = v > 0.f ? v : 0.f;
      }
    }
  }
}

// ---------------- host-side orchestration ----------------

extern "C" void kernel_launch(void* const* d_in, const int* in_sizes, int n_in,
                              void* d_out, int out_size, void* d_ws, size_t ws_size,
                              hipStream_t stream) {
  const float* f_atoms = (const float*)d_in[0];
  const float* f_bonds = (const float*)d_in[1];
  const int*   a2b     = (const int*)d_in[2];
  const int*   b2a     = (const int*)d_in[3];
  const int*   b2revb  = (const int*)d_in[4];
  const float* W_i     = (const float*)d_in[5];
  const float* W_h     = (const float*)d_in[6];
  const float* W_o     = (const float*)d_in[7];
  const float* b_o     = (const float*)d_in[8];
  float* out = (float*)d_out;

  char* ws = (char*)d_ws;
  size_t off = 0;
  auto salloc = [&](size_t bytes) -> char* {
    char* p = ws + off;
    off += (bytes + 255) & ~(size_t)255;
    return p;
  };
  float* inp  = (float*)salloc((size_t)NBq * HPAD * sizeof(float));   // 243.2 MB
  float* msg  = (float*)salloc((size_t)NBq * HPAD * sizeof(float));   // 243.2 MB
  float* amsg = (float*)salloc((size_t)NAq * HPAD * sizeof(float));   // 121.6 MB
  // time-multiplexed region: Abf (GEMM1 A) -> pre (GEMM2 A) -> ainp (GEMM3 A)
  char* region = salloc((size_t)NBq * KP_H * sizeof(__bf16));          // 128 MB
  __bf16* Abf  = (__bf16*)region;
  __bf16* pre  = (__bf16*)region;
  __bf16* ainp = (__bf16*)region;
  __bf16* Wi_sw = (__bf16*)salloc((size_t)NT_ALLOC * 5  * 32 * 16 * 2);
  __bf16* Wh_sw = (__bf16*)salloc((size_t)NT_ALLOC * 10 * 32 * 16 * 2);
  __bf16* Wo_sw = (__bf16*)salloc((size_t)NT_ALLOC * 14 * 32 * 16 * 2);

  // weight swizzles (tiny)
  {
    int tw = NT_ALLOC * 5 * 32;
    swizzle_w<<<(tw + 255) / 256, 256, 0, stream>>>(W_i, Wi_sw, BOND_FDIM, HDIM, 5);
    tw = NT_ALLOC * 10 * 32;
    swizzle_w<<<(tw + 255) / 256, 256, 0, stream>>>(W_h, Wh_sw, HDIM, HDIM, 10);
    tw = NT_ALLOC * 14 * 32;
    swizzle_w<<<(tw + 255) / 256, 256, 0, stream>>>(W_o, Wo_sw, ATOM_FDIM + HDIM, HDIM, 14);
  }

  // f_bonds -> bf16 padded A operand
  {
    long tot = (long)NBq * KP_BOND;
    cvt_pad_bf16<<<(int)((tot + 255) / 256), 256, 0, stream>>>(f_bonds, Abf, NBq, BOND_FDIM, KP_BOND);
  }

  const int mtilesB = NBq / 16;  // 12500
  const int mtilesA = NAq / 16;  // 6250
  dim3 gB((mtilesB + 7) / 8, 5);
  dim3 gA((mtilesA + 7) / 8, 5);

  // inp = f_bonds@W_i ; msg = relu(inp)
  gemm_wi<<<gB, 256, 0, stream>>>(Abf, Wi_sw, inp, msg, mtilesB, 5, KP_BOND);

  for (int it = 0; it < 2; ++it) {  // DEPTH-1 iterations
    int tg = NAq * (HPAD / 4);
    atom_gather_sum<<<(tg + 255) / 256, 256, 0, stream>>>(msg, a2b, amsg);
    int tb = NBq * (KP_H / 4);
    bond_prep<<<(tb + 255) / 256, 256, 0, stream>>>(amsg, msg, b2a, b2revb, pre);
    gemm_wh<<<gB, 256, 0, stream>>>(pre, Wh_sw, inp, msg, mtilesB, 10, KP_H);
  }

  // final gather + concat + output projection
  {
    int tg = NAq * (HPAD / 4);
    atom_gather_sum<<<(tg + 255) / 256, 256, 0, stream>>>(msg, a2b, amsg);
    long tc = (long)NAq * KP_O;
    build_ainput<<<(int)((tc + 255) / 256), 256, 0, stream>>>(f_atoms, amsg, ainp);
    gemm_wo<<<gA, 256, 0, stream>>>(ainp, Wo_sw, b_o, out, mtilesA, 14, KP_O);
  }
}
